// LogSigRNNGenerator_58128087384452
// MI455X (gfx1250) — matrix operand order repaired and checked by reading, not compiled, not run on hardware
//
#include <hip/hip_runtime.h>
#include <hip/hip_bf16.h>
#include <math.h>

typedef _Float16 v16h __attribute__((ext_vector_type(16)));
typedef _Float16 v8h  __attribute__((ext_vector_type(8)));
typedef float    v8f  __attribute__((ext_vector_type(8)));

#define GLOBAL_AS __attribute__((address_space(1)))

#define B_TOT   1024
#define T_NOISE 1000
#define HID     512
#define K1      544      // 518 padded to 17*32
#define NK1     17
#define NK2     16
#define LSW     32       // 6 logsig channels padded to one 32-wide K tile
#define WSTEPS  64
#define MROWS   16       // batch rows per workgroup

// workspace layout (bytes)
#define OFF_LS  0
#define SZ_LS   (WSTEPS * B_TOT * LSW * 2)
#define OFF_W1  (OFF_LS + SZ_LS)
#define SZ_W1   (HID * K1 * 2)
#define OFF_W2  (OFF_W1 + SZ_W1)
#define SZ_W2   (HID * HID * 2)
#define OFF_W3  (OFF_W2 + SZ_W2)
#define OFF_WO  (OFF_W3 + SZ_W2)
#define SZ_WO   (16 * HID * 2)

struct Sched { short ends[63]; };

// ---------------- preprocessing kernels ----------------

__global__ void fill_zero_f16(_Float16* p, int n) {
    int i = blockIdx.x * blockDim.x + threadIdx.x;
    if (i < n) p[i] = (_Float16)0.f;
}

__global__ void convert_weights(const float* __restrict__ W1, const float* __restrict__ W2,
                                const float* __restrict__ W3, const float* __restrict__ Wout,
                                _Float16* __restrict__ W1p, _Float16* __restrict__ W2p,
                                _Float16* __restrict__ W3p, _Float16* __restrict__ Wop) {
    int i = blockIdx.x * blockDim.x + threadIdx.x;
    if (i < HID * K1) {
        int n = i / K1, k = i % K1;
        W1p[i] = (k < 518) ? (_Float16)W1[n * 518 + k] : (_Float16)0.f;
    }
    if (i < HID * HID) {
        W2p[i] = (_Float16)W2[i];
        W3p[i] = (_Float16)W3[i];
    }
    if (i < 16 * HID) {
        int n = i / HID, k = i % HID;
        Wop[i] = (n < 3) ? (_Float16)Wout[n * HID + k] : (_Float16)0.f;
    }
}

// One thread per batch row: sequential scan of the Brownian path computing
// level-1 increments and Levy-area prefix sums. Window starts are always the
// last multiple-of-50 index <= the window end, so a single running snapshot
// (taken at t % 50 == 0) replaces C[:, starts].
__global__ void path_logsig(const float* __restrict__ z, _Float16* __restrict__ ls, Sched sc) {
    int b = blockIdx.x * blockDim.x + threadIdx.x;
    if (b >= B_TOT) return;
    const float sdt = sqrtf(1.0f / 999.0f);
    float bp0 = 0.f, bp1 = 0.f, bp2 = 0.f;
    float c01 = 0.f, c02 = 0.f, c12 = 0.f;
    float s01 = 0.f, s02 = 0.f, s12 = 0.f;
    int ke = 0;
    const float* zp = z + (size_t)b * T_NOISE * 3;
    for (int t = 1; t < T_NOISE; ++t) {
        float z0 = zp[t * 3 + 0] * sdt;
        float z1 = zp[t * 3 + 1] * sdt;
        float z2 = zp[t * 3 + 2] * sdt;
        float n0 = bp0 + z0, n1 = bp1 + z1, n2 = bp2 + z2;
        c01 += 0.5f * (bp0 * n1 - bp1 * n0);
        c02 += 0.5f * (bp0 * n2 - bp2 * n0);
        c12 += 0.5f * (bp1 * n2 - bp2 * n1);
        bp0 = n0; bp1 = n1; bp2 = n2;
        if (t % 50 == 0) { s01 = c01; s02 = c02; s12 = c12; }
        while (ke < 63 && (int)sc.ends[ke] == t) {
            _Float16* o = ls + ((size_t)(ke + 1) * B_TOT + b) * LSW;
            o[0] = (_Float16)bp0; o[1] = (_Float16)bp1; o[2] = (_Float16)bp2;
            o[3] = (_Float16)(c01 - s01);
            o[4] = (_Float16)(c02 - s02);
            o[5] = (_Float16)(c12 - s12);
            ++ke;
        }
    }
}

// ---------------- WMMA fragment loaders (gfx1250 wave32 layouts) ----------------

// Make a uniform pointer opaque so weight loads cannot be hoisted/CSE'd
// across the 64-step recurrence (which previously spilled ~1500 VGPRs of
// preloaded fragments to scratch).
__device__ __forceinline__ const _Float16* launder(const _Float16* p) {
    asm volatile("" : "+s"(p));
    return p;
}

// 16-bit A matrix 16x32: lanes 0-15 rows, 16 consecutive K from kb0 (+16 for upper half-wave)
__device__ __forceinline__ v16h load_a(const _Float16* src, int stride, int kbase, int lane) {
    int r  = lane & 15;
    int kb = kbase + ((lane >> 4) << 4);          // +16 for upper half-wave
    const _Float16* p = src + r * stride + kb;
    v8h lo = *(const v8h*)p;
    v8h hi = *(const v8h*)(p + 8);
    v16h a;
#pragma unroll
    for (int j = 0; j < 8; ++j) { a[j] = lo[j]; a[8 + j] = hi[j]; }
    return a;
}

// 16-bit B matrix 32x16 (B[k][n] = W[n][k]): lane holds one column, 16 consecutive K.
// Pin to address_space(1) via an inttoptr round-trip so the (laundered, hence
// generic) pointer still lowers to global_load_b128 instead of flat_load_b128
// (flat ties up the LDS path and counts against DScnt as well as LOADcnt).
__device__ __forceinline__ v16h load_b(const _Float16* W, int ldk,
                                       int ncol0, int kbase, int lane) {
    int col = ncol0 + (lane & 15);
    int kb  = kbase + ((lane >> 4) << 4);         // +16 for upper half-wave
    const _Float16* p = W + (size_t)col * ldk + kb;
    const GLOBAL_AS v16h* gp = (const GLOBAL_AS v16h*)(unsigned long long)(size_t)p;
    return *gp;
}

// One 16(M) x 64(N per wave) x K layer: each wave owns 4 N-tiles, A fragments
// shared across the 4 WMMAs per k-step. ACT: 0 = relu, 1 = tanh.
template <int ACT>
__device__ __forceinline__ void gemm_layer(const _Float16* in, int in_stride, int nK,
                                           const _Float16* Wg, int ldk,
                                           _Float16 (*outb)[HID], const float* __restrict__ bias,
                                           int lane, int wid) {
    v8f acc0 = {0.f,0.f,0.f,0.f,0.f,0.f,0.f,0.f};
    v8f acc1 = acc0, acc2 = acc0, acc3 = acc0;
#pragma unroll 2
    for (int kt = 0; kt < nK; ++kt) {
        v16h a = load_a(in, in_stride, kt * 32, lane);
        v16h bf0 = load_b(Wg, ldk, (wid * 4 + 0) * 16, kt * 32, lane);
        acc0 = __builtin_amdgcn_wmma_f32_16x16x32_f16(false, a, false, bf0, (short)0, acc0, false, false);
        v16h bf1 = load_b(Wg, ldk, (wid * 4 + 1) * 16, kt * 32, lane);
        acc1 = __builtin_amdgcn_wmma_f32_16x16x32_f16(false, a, false, bf1, (short)0, acc1, false, false);
        v16h bf2 = load_b(Wg, ldk, (wid * 4 + 2) * 16, kt * 32, lane);
        acc2 = __builtin_amdgcn_wmma_f32_16x16x32_f16(false, a, false, bf2, (short)0, acc2, false, false);
        v16h bf3 = load_b(Wg, ldk, (wid * 4 + 3) * 16, kt * 32, lane);
        acc3 = __builtin_amdgcn_wmma_f32_16x16x32_f16(false, a, false, bf3, (short)0, acc3, false, false);
    }
    // D layout: VGPR r -> row r + (lane>=16 ? 8 : 0), col = lane & 15
    int r0 = (lane >> 4) << 3;
    int rr = lane & 15;
    v8f accs[4] = {acc0, acc1, acc2, acc3};
#pragma unroll
    for (int j = 0; j < 4; ++j) {
        int col = (wid * 4 + j) * 16 + rr;
        float bv = bias[col];
#pragma unroll
        for (int r = 0; r < 8; ++r) {
            float v = accs[j][r] + bv;
            v = (ACT == 0) ? fmaxf(v, 0.f) : tanhf(v);
            outb[r0 + r][col] = (_Float16)v;
        }
    }
}

// ---------------- persistent RNN kernel: one block owns 16 batch rows ----------------

__global__ __launch_bounds__(256)
void rnn_kernel(const _Float16* __restrict__ ls,
                const _Float16* __restrict__ W1p, const _Float16* __restrict__ W2p,
                const _Float16* __restrict__ W3p, const _Float16* __restrict__ Wop,
                const float* __restrict__ b1, const float* __restrict__ b2,
                const float* __restrict__ b3, float* __restrict__ out,
                unsigned long long gates) {
    __shared__ __align__(32) _Float16 xbuf[MROWS][K1];   // [h | ls | pad]
    __shared__ __align__(32) _Float16 hb1[MROWS][HID];
    __shared__ __align__(32) _Float16 hb2[MROWS][HID];

    const int tid  = threadIdx.x;
    const int lane = tid & 31;
    const int wid  = tid >> 5;
    const int mbase = blockIdx.x * MROWS;

    // h0 = 0
    for (int i = tid; i < MROWS * HID; i += 256)
        xbuf[i / HID][i % HID] = (_Float16)0.f;
    __syncthreads();

#pragma unroll 1
    for (int t = 0; t < WSTEPS; ++t) {
        // fresh, opaque weight pointers each step: no cross-step hoisting
        const _Float16* W1t = launder(W1p);
        const _Float16* W2t = launder(W2p);
        const _Float16* W3t = launder(W3p);
        const _Float16* Wot = launder(Wop);

        // bring this step's (padded) log-signature into x columns [512, 544)
        for (int i = tid; i < MROWS * LSW; i += 256) {
            int r = i >> 5, c = i & 31;
            xbuf[r][HID + c] = ls[((size_t)t * B_TOT + (mbase + r)) * LSW + c];
        }
        __syncthreads();

        gemm_layer<0>(&xbuf[0][0], K1, NK1, W1t, K1, hb1, b1, lane, wid);   // relu
        __syncthreads();
        gemm_layer<0>(&hb1[0][0], HID, NK2, W2t, HID, hb2, b2, lane, wid);  // relu
        __syncthreads();
        gemm_layer<1>(&hb2[0][0], HID, NK2, W3t, HID, hb1, b3, lane, wid);  // tanh -> h3 in hb1
        __syncthreads();

        if (wid == 0) {
            // output projection: one 16x16 tile vs zero-padded Wout
            v8f acc = {0.f,0.f,0.f,0.f,0.f,0.f,0.f,0.f};
#pragma unroll 1
            for (int kt = 0; kt < NK2; ++kt) {
                v16h a  = load_a(&hb1[0][0], HID, kt * 32, lane);
                v16h bf = load_b(Wot, HID, 0, kt * 32, lane);
                acc = __builtin_amdgcn_wmma_f32_16x16x32_f16(false, a, false, bf, (short)0, acc, false, false);
            }
            int col = lane & 15;
            if (col < 3) {
                int r0 = (lane < 16) ? 0 : 8;
#pragma unroll
                for (int r = 0; r < 8; ++r)
                    out[((size_t)(mbase + r0 + r) * WSTEPS + t) * 3 + col] = acc[r];
            }
        } else if ((gates >> t) & 1ull) {
            // gated carry: h <- h3 (waves 1..7 copy while wave 0 does the projection)
            for (int i = tid - 32; i < MROWS * HID; i += 224)
                xbuf[i / HID][i % HID] = hb1[i / HID][i % HID];
        }
        __syncthreads();
    }
}

// ---------------- launch ----------------

extern "C" void kernel_launch(void* const* d_in, const int* in_sizes, int n_in,
                              void* d_out, int out_size, void* d_ws, size_t ws_size,
                              hipStream_t stream) {
    (void)in_sizes; (void)n_in; (void)out_size; (void)ws_size;
    const float* z    = (const float*)d_in[0];
    const float* W1   = (const float*)d_in[1];
    const float* b1   = (const float*)d_in[2];
    const float* W2   = (const float*)d_in[3];
    const float* b2   = (const float*)d_in[4];
    const float* W3   = (const float*)d_in[5];
    const float* b3   = (const float*)d_in[6];
    const float* Wout = (const float*)d_in[7];
    float* out = (float*)d_out;

    char* ws = (char*)d_ws;
    _Float16* ls  = (_Float16*)(ws + OFF_LS);
    _Float16* W1p = (_Float16*)(ws + OFF_W1);
    _Float16* W2p = (_Float16*)(ws + OFF_W2);
    _Float16* W3p = (_Float16*)(ws + OFF_W3);
    _Float16* Wop = (_Float16*)(ws + OFF_WO);

    // Replicate the static numpy schedule (double precision, searchsorted 'right').
    double tb[1000], tu[20], tt[64];
    for (int i = 0; i < 1000; ++i) tb[i] = (double)i / 999.0;
    for (int j = 0; j < 20; ++j)  tu[j] = tb[50 * j];
    for (int i = 0; i < 64; ++i)  tt[i] = (double)i / 63.0;

    Sched sc;
    int uoft[63];
    for (int k = 0; k < 63; ++k) {
        double v = tt[k + 1];
        int it = 0; while (it < 1000 && tb[it] <= v) ++it; it -= 1;
        int iu = 0; while (iu < 20 && tu[iu] <= v) ++iu; iu -= 1;
        if (iu < 0) iu = 0;
        sc.ends[k] = (short)it;
        uoft[k] = iu;
    }
    double utimes[64]; int nu = 0, last = -1;
    for (int k = 0; k < 63; ++k)
        if (uoft[k] != last) { utimes[nu++] = tu[uoft[k]]; last = uoft[k]; }
    utimes[nu++] = tt[63];
    unsigned long long gates = 0; int qi = 0;
    for (int i = 0; i < 64; ++i)
        if (qi < nu && tt[i] >= utimes[qi]) { ++qi; gates |= (1ull << i); }

    fill_zero_f16<<<(SZ_LS / 2 + 255) / 256, 256, 0, stream>>>(ls, SZ_LS / 2);
    convert_weights<<<(HID * K1 + 255) / 256, 256, 0, stream>>>(W1, W2, W3, Wout, W1p, W2p, W3p, Wop);
    path_logsig<<<(B_TOT + 255) / 256, 256, 0, stream>>>(z, ls, sc);
    rnn_kernel<<<B_TOT / MROWS, 256, 0, stream>>>(ls, W1p, W2p, W3p, Wop, b1, b2, b3, out, gates);
}
